// VSPN_42726334660744
// MI455X (gfx1250) — matrix operand order repaired and checked by reading, not compiled, not run on hardware
//
#include <hip/hip_runtime.h>
#include <hip/hip_bf16.h>

// ---------------------------------------------------------------------------
// MI455X (gfx1250) implementation of the dual-graph GatedGraphConv network.
//
// Strategy (see analysis): all GEMMs run on v_wmma_f32_16x16x32_bf16 with
// fp32 accumulation. Weights are packed once per launch into bf16 WMMA
// B-fragment layout; activations stay fp32 in memory and are converted to
// bf16 fragments in registers. The GRU (two N x 128 x 384 GEMMs + gates) is
// fused into a single kernel so gi/gh (2 x 150 MB per step) are never
// materialized. Scatter/pool use fp32 global atomics.
// ---------------------------------------------------------------------------

typedef __attribute__((ext_vector_type(16))) __bf16 v16bf;
typedef __attribute__((ext_vector_type(8)))  float  v8f;

static constexpr int NN  = 100000;  // nodes
static constexpr int EE  = 600000;  // edges
static constexpr int GG  = 64;      // graphs
static constexpr int HH  = 128;     // hidden
static constexpr int ST  = 4;       // GGC steps

// ---------------------------------------------------------------------------
// A-fragment loaders: 16x32 bf16 A tile for v_wmma_f32_16x16x32_bf16.
// Lane L (<16) holds row L, K offsets {0..7, 16..23}; lanes >=16 hold row
// L-16, K offsets {8..15, 24..31}  (ISA 7.12.2, 16-bit A 16x32 layout).
// ---------------------------------------------------------------------------
__device__ __forceinline__ v16bf a_frag_pred(const float* __restrict__ A, int lda,
                                             int row0, int k0, int K) {
  int ln = threadIdx.x & 31;
  const float* base = A + (size_t)(row0 + (ln & 15)) * lda;
  int kb = k0 + ((ln >> 4) << 3);
  v16bf a;
#pragma unroll
  for (int e = 0; e < 16; ++e) {
    int k = kb + (e < 8 ? e : e + 8);
    a[e] = (k < K) ? (__bf16)base[k] : (__bf16)0.0f;
  }
  return a;
}

__device__ __forceinline__ v16bf a_frag_128(const float* __restrict__ A,
                                            int row0, int k0) {
  int ln = threadIdx.x & 31;
  const float* p = A + (size_t)(row0 + (ln & 15)) * 128 + k0 + ((ln >> 4) << 3);
  float4 f0 = *(const float4*)(p);
  float4 f1 = *(const float4*)(p + 4);
  float4 f2 = *(const float4*)(p + 16);
  float4 f3 = *(const float4*)(p + 20);
  v16bf a;
  a[0]  = (__bf16)f0.x; a[1]  = (__bf16)f0.y; a[2]  = (__bf16)f0.z; a[3]  = (__bf16)f0.w;
  a[4]  = (__bf16)f1.x; a[5]  = (__bf16)f1.y; a[6]  = (__bf16)f1.z; a[7]  = (__bf16)f1.w;
  a[8]  = (__bf16)f2.x; a[9]  = (__bf16)f2.y; a[10] = (__bf16)f2.z; a[11] = (__bf16)f2.w;
  a[12] = (__bf16)f3.x; a[13] = (__bf16)f3.y; a[14] = (__bf16)f3.z; a[15] = (__bf16)f3.w;
  return a;
}

// ---------------------------------------------------------------------------
// Weight packing: fp32 weight -> bf16 B fragments.
// B tile is 32x16 (K x N). Same K order as the A loaders: lanes 0-15 hold
// K={0..7,16..23} (column = lane), lanes 16-31 hold K={8..15,24..31} for column lane-16; the
// 16 half-values per lane walk K in two runs of 8.  One wave packs one
// (colTile, kStep) fragment: 32 lanes x 16 bf16 -> 1 KB.
// transpose==0: B[k,n] = W[k*ncols + n]    (W is K x ncols row-major)
// transpose==1: B[k,n] = W[n*K + k]        (W is ncols x K row-major, used as W^T)
// Fragments stored as out[(tile*ksteps + ks)*512 + lane*16 + e].
// ---------------------------------------------------------------------------
__global__ void pack_b_kernel(const float* __restrict__ W, __bf16* __restrict__ out,
                              int K, int ncols, int ksteps, int transpose) {
  int wid  = blockIdx.x * (blockDim.x >> 5) + (threadIdx.x >> 5);
  int ln   = threadIdx.x & 31;
  int nT   = ncols >> 4;
  if (wid >= nT * ksteps) return;
  int tile = wid / ksteps;
  int ks   = wid % ksteps;
  int n    = tile * 16 + (ln & 15);
  int kb   = ks * 32 + ((ln >> 4) << 3);  // +8 for upper lane half
  __bf16* dst = out + (size_t)wid * 512 + ln * 16;
#pragma unroll
  for (int e = 0; e < 16; ++e) {
    int k = kb + (e < 8 ? e : e + 8);
    float f = 0.0f;
    if (k < K) f = transpose ? W[(size_t)n * K + k] : W[(size_t)k * ncols + n];
    dst[e] = (__bf16)f;
  }
}

__global__ void fill0_kernel(float* __restrict__ p, size_t n) {
  size_t i = (size_t)blockIdx.x * blockDim.x + threadIdx.x;
  if (i < n) p[i] = 0.0f;
}

// ---------------------------------------------------------------------------
// Generic strip GEMM: C[M x 128] = act(A[M x K] * Bpacked + bias).
// One wave owns a 16-row strip and all 8 column tiles (A fragment reused 8x).
// EXACT=true -> K==128, lda==128 vectorized fragment loads.
// ---------------------------------------------------------------------------
template <bool EXACT>
__global__ void gemm_h128_kernel(const float* __restrict__ A, int M, int K, int lda,
                                 int ksteps, const __bf16* __restrict__ Bp,
                                 const float* __restrict__ bias,
                                 float* __restrict__ C, int relu) {
  int strip = blockIdx.x * (blockDim.x >> 5) + (threadIdx.x >> 5);
  if (strip * 16 >= M) return;          // wave-uniform: EXEC stays all-ones
  int ln = threadIdx.x & 31;
  v8f acc[8] = {};
  for (int ks = 0; ks < ksteps; ++ks) {
    v16bf a = EXACT ? a_frag_128(A, strip * 16, ks * 32)
                    : a_frag_pred(A, lda, strip * 16, ks * 32, K);
#pragma unroll
    for (int t = 0; t < 8; ++t) {
      v16bf b = *(const v16bf*)(Bp + ((size_t)t * ksteps + ks) * 512 + ln * 16);
      acc[t] = __builtin_amdgcn_wmma_f32_16x16x32_bf16(
          false, a, false, b, (short)0, acc[t], false, false);
    }
  }
  int mb = strip * 16 + ((ln >> 4) << 3);  // M = v + (lane>=16 ? 8 : 0)
  int nc = ln & 15;
#pragma unroll
  for (int t = 0; t < 8; ++t) {
    int n = t * 16 + nc;
    float bv = bias ? bias[n] : 0.0f;
#pragma unroll
    for (int v = 0; v < 8; ++v) {
      float val = acc[t][v] + bv;
      if (relu) val = fmaxf(val, 0.0f);
      C[(size_t)(mb + v) * 128 + n] = val;
    }
  }
}

// ---------------------------------------------------------------------------
// Edge scatter: agg[dst] += m[src].  One wave per edge, float4 per lane.
// ---------------------------------------------------------------------------
__global__ void scatter_kernel(const float* __restrict__ m, const int* __restrict__ edge,
                               float* __restrict__ agg, int E) {
  int tid  = blockIdx.x * blockDim.x + threadIdx.x;
  int e    = tid >> 5;
  int ln   = threadIdx.x & 31;
  if (e >= E) return;
  int src = edge[e];
  int dst = edge[E + e];
  float4 v = *(const float4*)(m + (size_t)src * 128 + ln * 4);
  float* o = agg + (size_t)dst * 128 + ln * 4;
  atomicAdd(o + 0, v.x);
  atomicAdd(o + 1, v.y);
  atomicAdd(o + 2, v.z);
  atomicAdd(o + 3, v.w);
}

// ---------------------------------------------------------------------------
// Fused GRU: for one 16x16 block of the new hidden state, compute the r/z/n
// column tiles of gi = agg*Wih^T and gh = h*Whh^T (6 WMMA accumulators),
// then apply the gate math and write h_next. Never materializes gi/gh.
// grid = (strips, 8 column groups).
// ---------------------------------------------------------------------------
__global__ void gru_fused_kernel(const float* __restrict__ agg, const float* __restrict__ h,
                                 const __bf16* __restrict__ Bih, const __bf16* __restrict__ Bhh,
                                 const float* __restrict__ b_ih, const float* __restrict__ b_hh,
                                 float* __restrict__ h_out, int M) {
  int strip = blockIdx.x * (blockDim.x >> 5) + (threadIdx.x >> 5);
  if (strip * 16 >= M) return;
  int g  = blockIdx.y;        // 0..7 -> hidden columns g*16..g*16+15
  int ln = threadIdx.x & 31;
  v8f aI[3] = {};
  v8f aH[3] = {};
  for (int ks = 0; ks < 4; ++ks) {
    v16bf fa = a_frag_128(agg, strip * 16, ks * 32);
    v16bf fh = a_frag_128(h,   strip * 16, ks * 32);
#pragma unroll
    for (int c = 0; c < 3; ++c) {
      int tile = g + c * 8;  // column tile within the 384-wide gate matrix
      v16bf bi = *(const v16bf*)(Bih + ((size_t)tile * 4 + ks) * 512 + ln * 16);
      aI[c] = __builtin_amdgcn_wmma_f32_16x16x32_bf16(
          false, fa, false, bi, (short)0, aI[c], false, false);
      v16bf bh = *(const v16bf*)(Bhh + ((size_t)tile * 4 + ks) * 512 + ln * 16);
      aH[c] = __builtin_amdgcn_wmma_f32_16x16x32_bf16(
          false, fh, false, bh, (short)0, aH[c], false, false);
    }
  }
  int mb = strip * 16 + ((ln >> 4) << 3);
  int nc = g * 16 + (ln & 15);
  float bir = b_ih[nc], biz = b_ih[128 + nc], bin = b_ih[256 + nc];
  float bhr = b_hh[nc], bhz = b_hh[128 + nc], bhn = b_hh[256 + nc];
#pragma unroll
  for (int v = 0; v < 8; ++v) {
    float ir = aI[0][v] + bir, iz = aI[1][v] + biz, in_ = aI[2][v] + bin;
    float hr = aH[0][v] + bhr, hz = aH[1][v] + bhz, hn  = aH[2][v] + bhn;
    float r = 1.0f / (1.0f + __expf(-(ir + hr)));
    float z = 1.0f / (1.0f + __expf(-(iz + hz)));
    float n = tanhf(in_ + r * hn);
    float hold = h[(size_t)(mb + v) * 128 + nc];
    h_out[(size_t)(mb + v) * 128 + nc] = (1.0f - z) * n + z * hold;
  }
}

// ---------------------------------------------------------------------------
// ReLU + segment-sum pool: one wave per node, float4 per lane; lane 0 counts.
// ---------------------------------------------------------------------------
__global__ void pool_kernel(const float* __restrict__ h, const int* __restrict__ batch,
                            float* __restrict__ sums, float* __restrict__ cnt, int Nn) {
  int tid  = blockIdx.x * blockDim.x + threadIdx.x;
  int node = tid >> 5;
  int ln   = threadIdx.x & 31;
  if (node >= Nn) return;
  int b = batch[node];
  float4 v = *(const float4*)(h + (size_t)node * 128 + ln * 4);
  float* o = sums + (size_t)b * 128 + ln * 4;
  atomicAdd(o + 0, fmaxf(v.x, 0.0f));
  atomicAdd(o + 1, fmaxf(v.y, 0.0f));
  atomicAdd(o + 2, fmaxf(v.z, 0.0f));
  atomicAdd(o + 3, fmaxf(v.w, 0.0f));
  if (ln == 0) atomicAdd(&cnt[b], 1.0f);
}

__global__ void finalize_g_kernel(const float* __restrict__ sb, const float* __restrict__ cb,
                                  const float* __restrict__ sv, const float* __restrict__ cv,
                                  float* __restrict__ gvec) {
  int g = blockIdx.x;
  int c = threadIdx.x;  // 0..255
  float val;
  if (c < 128) val = sb[g * 128 + c] / fmaxf(cb[g], 1.0f);
  else         val = sv[g * 128 + (c - 128)] / fmaxf(cv[g], 1.0f);
  gvec[g * 256 + c] = val;
}

// out[g] = softplus(relu(gvec[g] @ W1 + b1) @ W2 + b2).  ~8 MFLOP total: VALU.
__global__ void mlp_kernel(const float* __restrict__ gvec, const float* __restrict__ W1,
                           const float* __restrict__ b1, const float* __restrict__ W2,
                           const float* __restrict__ b2, float* __restrict__ out) {
  __shared__ float grow[256];
  __shared__ float red[256];
  int g = blockIdx.x;
  int c = threadIdx.x;
  grow[c] = gvec[g * 256 + c];
  __syncthreads();
  float acc = b1[c];
  for (int k = 0; k < 256; ++k) acc += grow[k] * W1[k * 256 + c];
  red[c] = fmaxf(acc, 0.0f) * W2[c];
  __syncthreads();
  for (int s = 128; s > 0; s >>= 1) {
    if (c < s) red[c] += red[c + s];
    __syncthreads();
  }
  if (c == 0) {
    float x = red[0] + b2[0];
    out[g] = (x > 20.0f) ? x : log1pf(__expf(x));
  }
}

// ---------------------------------------------------------------------------
extern "C" void kernel_launch(void* const* d_in, const int* in_sizes, int n_in,
                              void* d_out, int out_size, void* d_ws, size_t ws_size,
                              hipStream_t stream) {
  (void)in_sizes; (void)n_in; (void)out_size; (void)ws_size;

  const float* x_b      = (const float*)d_in[0];
  const int*   ei_b     = (const int*)  d_in[1];
  const int*   batch_b  = (const int*)  d_in[2];
  const float* x_v      = (const float*)d_in[3];
  const int*   ei_v     = (const int*)  d_in[4];
  const int*   batch_v  = (const int*)  d_in[5];
  const float* W_emb_b  = (const float*)d_in[6];   // [90,128]
  const float* W_emb_v  = (const float*)d_in[7];   // [20,128]
  const float* ggc_w    = (const float*)d_in[8];   // [4,128,128]
  const float* gru_w_ih = (const float*)d_in[9];   // [384,128]
  const float* gru_w_hh = (const float*)d_in[10];  // [384,128]
  const float* gru_b_ih = (const float*)d_in[11];  // [384]
  const float* gru_b_hh = (const float*)d_in[12];  // [384]
  const float* W1       = (const float*)d_in[13];  // [256,256]
  const float* b1       = (const float*)d_in[14];  // [256]
  const float* W2       = (const float*)d_in[15];  // [256,1]
  const float* b2       = (const float*)d_in[16];  // [1]
  float* out = (float*)d_out;

  // -------- workspace carve-up (all offsets 256B aligned) --------
  char* wsb = (char*)d_ws;
  size_t off = 0;
  auto carve = [&](size_t bytes) -> char* {
    char* p = wsb + off;
    off = (off + bytes + 255) & ~(size_t)255;
    return p;
  };
  const size_t NBf = (size_t)NN * HH;           // floats per activation buffer
  float* hbuf = (float*)carve(NBf * 4);         // current hidden / embeddings
  float* htmp = (float*)carve(NBf * 4);         // ping-pong hidden
  float* mbuf = (float*)carve(NBf * 4);         // message GEMM output
  float* aggb = (float*)carve(NBf * 4);         // scatter accumulator
  __bf16* pEmbB = (__bf16*)carve((size_t)8 * 3 * 512 * 2);
  __bf16* pEmbV = (__bf16*)carve((size_t)8 * 1 * 512 * 2);
  __bf16* pGgc  = (__bf16*)carve((size_t)ST * 8 * 4 * 512 * 2);
  __bf16* pIh   = (__bf16*)carve((size_t)24 * 4 * 512 * 2);
  __bf16* pHh   = (__bf16*)carve((size_t)24 * 4 * 512 * 2);
  float* sums = (float*)carve((size_t)2 * GG * HH * 4);
  float* cnts = (float*)carve((size_t)2 * GG * 4);
  float* gvec = (float*)carve((size_t)GG * 256 * 4);

  // -------- pack weights to bf16 WMMA B-fragments --------
  auto pack = [&](const float* W, __bf16* dst, int K, int ncols, int ksteps, int tr) {
    int waves = (ncols >> 4) * ksteps;
    pack_b_kernel<<<(waves + 3) / 4, 128, 0, stream>>>(W, dst, K, ncols, ksteps, tr);
  };
  pack(W_emb_b, pEmbB, 90, 128, 3, 0);
  pack(W_emb_v, pEmbV, 20, 128, 1, 0);
  for (int s = 0; s < ST; ++s)
    pack(ggc_w + (size_t)s * HH * HH, pGgc + (size_t)s * 8 * 4 * 512, 128, 128, 4, 0);
  pack(gru_w_ih, pIh, 128, 384, 4, 1);  // B = W_ih^T
  pack(gru_w_hh, pHh, 128, 384, 4, 1);  // B = W_hh^T

  // -------- zero pooling accumulators --------
  {
    size_t n = (size_t)2 * GG * HH + 2 * GG;
    fill0_kernel<<<(unsigned)((n + 255) / 256), 256, 0, stream>>>(sums, n);  // sums+cnts contiguous
  }

  const int strips      = NN / 16;                 // 6250 (NN % 16 == 0)
  const int gemmBlocks  = (strips + 3) / 4;        // 4 waves / block
  const dim3 gruGrid(gemmBlocks, 8);
  const int scatBlocks  = (EE * 32 + 255) / 256;
  const int poolBlocks  = (NN * 32 + 255) / 256;
  const int fillBlocks  = (int)((NBf + 255) / 256);

  const float* Xg[2]   = {x_b, x_v};
  const int*   Eg[2]   = {ei_b, ei_v};
  const int*   Bg[2]   = {batch_b, batch_v};
  const int    Kemb[2] = {90, 20};
  const int    KSe[2]  = {3, 1};
  const __bf16* Pe[2]  = {pEmbB, pEmbV};

  for (int gsel = 0; gsel < 2; ++gsel) {
    // embedding: h = relu(x @ W_emb)
    gemm_h128_kernel<false><<<gemmBlocks, 128, 0, stream>>>(
        Xg[gsel], NN, Kemb[gsel], Kemb[gsel], KSe[gsel], Pe[gsel], nullptr, hbuf, 1);

    float* cur = hbuf;
    float* nxt = htmp;
    for (int s = 0; s < ST; ++s) {
      // m = h @ ggc_w[s]
      gemm_h128_kernel<true><<<gemmBlocks, 128, 0, stream>>>(
          cur, NN, 128, 128, 4, pGgc + (size_t)s * 8 * 4 * 512, nullptr, mbuf, 0);
      // agg = segment_sum(m[src] -> dst)
      fill0_kernel<<<fillBlocks, 256, 0, stream>>>(aggb, NBf);
      scatter_kernel<<<scatBlocks, 256, 0, stream>>>(mbuf, Eg[gsel], aggb, EE);
      // h = GRU(agg, h)   (fused dual-GEMM + gates)
      gru_fused_kernel<<<gruGrid, 128, 0, stream>>>(
          aggb, cur, pIh, pHh, gru_b_ih, gru_b_hh, nxt, NN);
      float* t = cur; cur = nxt; nxt = t;
    }
    // mean-pool relu(h) per graph
    pool_kernel<<<poolBlocks, 256, 0, stream>>>(
        cur, Bg[gsel], sums + (size_t)gsel * GG * HH, cnts + (size_t)gsel * GG, NN);
  }

  finalize_g_kernel<<<GG, 256, 0, stream>>>(
      sums, cnts, sums + (size_t)GG * HH, cnts + GG, gvec);
  mlp_kernel<<<GG, 256, 0, stream>>>(gvec, W1, b1, W2, b2, out);
}
